// _FourierLayer_73194832659109
// MI455X (gfx1250) — matrix operand order repaired and checked by reading, not compiled, not run on hardware
//
// CDNA5 (gfx1250) Fourier Neural Operator layer.
// out = irfft2( kernel * rfft2(points)[20x20 modes] ) + points @ w_lin^T + points
// Implemented as 5 truncated-DFT GEMM stages, all matrix math on
// v_wmma_f32_16x16x32_f16 (f16 inputs, f32 accumulate). Complex arithmetic is
// folded into the K dimension ([re|im] concatenation), twiddles generated with
// exact integer-mod phases + cospif/sinpif. ~32 GFLOP vs ~0.9 GB HBM traffic
// => memory-bound; design minimizes global traffic (points read 2x, out 1x).
#include <hip/hip_runtime.h>

constexpr int Bn = 16, Hn = 256, Wn = 256, Cn = 64;
constexpr int M1n = 20, M2n = 20;

typedef __attribute__((ext_vector_type(16))) _Float16 v16h;
typedef __attribute__((ext_vector_type(8)))  _Float16 v8h;
typedef __attribute__((ext_vector_type(8)))  float    v8f;

__device__ __forceinline__ v8f wmma16(v16h a, v16h b, v8f c) {
  // (neg_a, A, neg_b, B, c_mod, C, reuse_a, reuse_b)
  return __builtin_amdgcn_wmma_f32_16x16x32_f16(false, a, false, b, (short)0, c,
                                                false, false);
}

// A fragment (16x32 f16, row-major tile in LDS, row stride ldk halfs).
// Lane layout (ISA 7.12.2): lanes 0-15 row M=lane, K = {0..7, 16..23};
// lanes 16-31 row M=lane-16, K = {8..15, 24..31}.
__device__ __forceinline__ v16h frag_a(const _Float16* tile, int ldk, int lane) {
  const int r = lane & 15, lg = lane >> 4;
  const _Float16* p = tile + r * ldk + 8 * lg;
  v8h lo = *(const v8h*)(p);        // K = 8g .. 8g+7
  v8h hi = *(const v8h*)(p + 16);   // K = 8g+16 .. 8g+23
  return __builtin_shufflevector(lo, hi, 0, 1, 2, 3, 4, 5, 6, 7,
                                 8, 9, 10, 11, 12, 13, 14, 15);
}

// B fragment (32x16 f16). tileT is stored transposed [N][K] in LDS.
// Lanes 0-15: column N=lane, K = {0..7, 16..23}; lanes 16-31: N=lane-16, K = {8..15, 24..31}.
__device__ __forceinline__ v16h frag_b(const _Float16* tileT, int ldk, int lane) {
  const int n = lane & 15, lg = lane >> 4;
  const _Float16* p = tileT + n * ldk + 8 * lg;
  v8h lo = *(const v8h*)(p);
  v8h hi = *(const v8h*)(p + 16);
  return __builtin_shufflevector(lo, hi, 0, 1, 2, 3, 4, 5, 6, 7,
                                 8, 9, 10, 11, 12, 13, 14, 15);
}

// ---------------------------------------------------------------------------
// Stage 1: truncated DFT along W.  t[b,h,y,c] = sum_w P[b,h,w,c] e^{-2pi i yw/W}
// One workgroup per (b,h). GEMM: M=32(pad of 20 y), N=64(c), K=256(w).
// ---------------------------------------------------------------------------
__global__ __launch_bounds__(128) void k1_dft_w(const float* __restrict__ points,
                                                float* __restrict__ t_re,
                                                float* __restrict__ t_im) {
  __shared__ __align__(64) _Float16 ldsBT[64][32];   // P^T chunk [c][w-local]
  __shared__ __align__(64) _Float16 ldsAre[32][32];  // cos twiddles
  __shared__ __align__(64) _Float16 ldsAim[32][32];  // -sin twiddles
  const int bh = blockIdx.x;
  const int tid = threadIdx.x, wid = tid >> 5, lane = tid & 31;
  const size_t pbase = (size_t)bh * Wn * Cn;
  v8f accRe[2] = {}, accIm[2] = {};

  for (int kb = 0; kb < 8; ++kb) {
    const int k0 = kb * 32;
    if (kb < 7)  // global_prefetch_b8 of next K-chunk
      __builtin_prefetch(points + pbase + (size_t)(k0 + 32) * Cn, 0, 1);
    for (int idx = tid; idx < 2048; idx += 128) {
      int c = idx & 63, wl = idx >> 6;
      ldsBT[c][wl] = (_Float16)points[pbase + (size_t)(k0 + wl) * Cn + c];
    }
    for (int idx = tid; idx < 1024; idx += 128) {
      int y = idx >> 5, wl = idx & 31;
      float cre = 0.f, cim = 0.f;
      if (y < M2n) {
        int r = (y * (k0 + wl)) & 255;              // phase mod W, exact
        float a = (float)r * (1.0f / 128.0f);       // angle = pi*a
        cre = cospif(a);
        cim = -sinpif(a);
      }
      ldsAre[y][wl] = (_Float16)cre;
      ldsAim[y][wl] = (_Float16)cim;
    }
    __syncthreads();
    v16h bfrag = frag_b(&ldsBT[wid * 16][0], 32, lane);
    for (int mt = 0; mt < 2; ++mt) {
      v16h are = frag_a(&ldsAre[mt * 16][0], 32, lane);
      v16h aim = frag_a(&ldsAim[mt * 16][0], 32, lane);
      accRe[mt] = wmma16(are, bfrag, accRe[mt]);
      accIm[mt] = wmma16(aim, bfrag, accIm[mt]);
    }
    __syncthreads();
  }
  const int lg = lane >> 4, nl = lane & 15;
  const int c = wid * 16 + nl;
  for (int mt = 0; mt < 2; ++mt)
    for (int e = 0; e < 8; ++e) {
      int y = mt * 16 + e + 8 * lg;
      if (y < M2n) {
        size_t o = ((size_t)bh * M2n + y) * Cn + c;
        t_re[o] = accRe[mt][e];
        t_im[o] = accIm[mt][e];
      }
    }
}

// ---------------------------------------------------------------------------
// Stage 2: DFT along H onto 20 modes (freq f = x-10), complex input.
// One workgroup per (b,ky). GEMM: M=32(pad x), N=64(c), K=512 ([t_re|t_im]).
// ---------------------------------------------------------------------------
__global__ __launch_bounds__(128) void k2_dft_h(const float* __restrict__ t_re,
                                                const float* __restrict__ t_im,
                                                float* __restrict__ m_re,
                                                float* __restrict__ m_im) {
  __shared__ __align__(64) _Float16 ldsBT[64][32];
  __shared__ __align__(64) _Float16 ldsAre[32][32];
  __shared__ __align__(64) _Float16 ldsAim[32][32];
  const int b = blockIdx.x / M2n, ky = blockIdx.x % M2n;
  const int tid = threadIdx.x, wid = tid >> 5, lane = tid & 31;
  v8f accRe[2] = {}, accIm[2] = {};

  for (int kb = 0; kb < 16; ++kb) {
    const int half = kb >> 3;            // 0: t_re rows, 1: t_im rows
    const int hbase = (kb & 7) * 32;
    const float* src = half ? t_im : t_re;
    for (int idx = tid; idx < 2048; idx += 128) {
      int c = idx & 63, kk = idx >> 6;
      ldsBT[c][kk] =
          (_Float16)src[((size_t)(b * Hn + hbase + kk) * M2n + ky) * Cn + c];
    }
    for (int idx = tid; idx < 1024; idx += 128) {
      int x = idx >> 5, kk = idx & 31;
      float vre = 0.f, vim = 0.f;
      if (x < M1n) {
        int f = x - 10;                   // mode frequency (idx1 mapping)
        int r = (f * (hbase + kk)) & 255; // two's-complement & == mod 256
        float a = (float)r * (1.0f / 128.0f);
        float c0 = cospif(a), s0 = sinpif(a);
        vre = half ? s0 : c0;             // modes_re = c0*t_re + s0*t_im
        vim = half ? c0 : -s0;            // modes_im = -s0*t_re + c0*t_im
      }
      ldsAre[x][kk] = (_Float16)vre;
      ldsAim[x][kk] = (_Float16)vim;
    }
    __syncthreads();
    v16h bfrag = frag_b(&ldsBT[wid * 16][0], 32, lane);
    for (int mt = 0; mt < 2; ++mt) {
      v16h are = frag_a(&ldsAre[mt * 16][0], 32, lane);
      v16h aim = frag_a(&ldsAim[mt * 16][0], 32, lane);
      accRe[mt] = wmma16(are, bfrag, accRe[mt]);
      accIm[mt] = wmma16(aim, bfrag, accIm[mt]);
    }
    __syncthreads();
  }
  const int lg = lane >> 4, nl = lane & 15;
  const int c = wid * 16 + nl;
  for (int mt = 0; mt < 2; ++mt)
    for (int e = 0; e < 8; ++e) {
      int x = mt * 16 + e + 8 * lg;
      if (x < M1n) {
        size_t o = (((size_t)b * M1n + x) * M2n + ky) * Cn + c;
        m_re[o] = accRe[mt][e];
        m_im[o] = accIm[mt][e];
      }
    }
}

// ---------------------------------------------------------------------------
// Stage 3: per-mode complex 64x64 mixing, batched over B.
// One workgroup per (kx,ky). GEMM: M=16(b), N=64(i), K=128 ([m_re|m_im] over j).
// ---------------------------------------------------------------------------
__global__ __launch_bounds__(128) void k3_modes(const float* __restrict__ m_re,
                                                const float* __restrict__ m_im,
                                                const float* __restrict__ ker_re,
                                                const float* __restrict__ ker_im,
                                                float* __restrict__ o_re,
                                                float* __restrict__ o_im) {
  __shared__ __align__(64) _Float16 ldsA[16][32];
  __shared__ __align__(64) _Float16 ldsBTre[64][32];
  __shared__ __align__(64) _Float16 ldsBTim[64][32];
  const int kx = blockIdx.x / M2n, ky = blockIdx.x % M2n;
  const int tid = threadIdx.x, wid = tid >> 5, lane = tid & 31;
  const size_t kbase = (size_t)(kx * M2n + ky) * Cn * Cn;
  v8f accRe = {}, accIm = {};

  for (int ks = 0; ks < 4; ++ks) {
    const int jh = ks >> 1;             // 0: first j-half (re-path), 1: second
    const int j0 = (ks & 1) * 32;
    const float* msrc = jh ? m_im : m_re;
    for (int idx = tid; idx < 512; idx += 128) {
      int brow = idx >> 5, col = idx & 31;
      ldsA[brow][col] =
          (_Float16)msrc[(((size_t)brow * M1n + kx) * M2n + ky) * Cn + j0 + col];
    }
    for (int idx = tid; idx < 2048; idx += 128) {
      int i = idx >> 5, col = idx & 31;
      float kr = ker_re[kbase + (size_t)i * Cn + j0 + col];
      float ki = ker_im[kbase + (size_t)i * Cn + j0 + col];
      // out_re = m_re@Kr^T - m_im@Ki^T ; out_im = m_re@Ki^T + m_im@Kr^T
      ldsBTre[i][col] = (_Float16)(jh ? -ki : kr);
      ldsBTim[i][col] = (_Float16)(jh ? kr : ki);
    }
    __syncthreads();
    v16h afrag = frag_a(&ldsA[0][0], 32, lane);
    v16h bre = frag_b(&ldsBTre[wid * 16][0], 32, lane);
    v16h bim = frag_b(&ldsBTim[wid * 16][0], 32, lane);
    accRe = wmma16(afrag, bre, accRe);
    accIm = wmma16(afrag, bim, accIm);
    __syncthreads();
  }
  const int lg = lane >> 4, nl = lane & 15;
  const int c = wid * 16 + nl;
  for (int e = 0; e < 8; ++e) {
    int brow = e + 8 * lg;
    size_t o = (((size_t)brow * M1n + kx) * M2n + ky) * Cn + c;
    o_re[o] = accRe[e];
    o_im[o] = accIm[e];
  }
}

// ---------------------------------------------------------------------------
// Stage 4: inverse DFT along H (includes 1/H).  One workgroup per (b,ky).
// GEMM: M=256(h), N=64(c), K=64 ([m2_re|m2_im] over kx, padded 20->32 each).
// ---------------------------------------------------------------------------
__global__ __launch_bounds__(128) void k4_idft_h(const float* __restrict__ m2_re,
                                                 const float* __restrict__ m2_im,
                                                 float* __restrict__ g_re,
                                                 float* __restrict__ g_im) {
  __shared__ __align__(64) _Float16 ldsBT[64][64];
  __shared__ __align__(64) _Float16 ldsAre[256][64];
  __shared__ __align__(64) _Float16 ldsAim[256][64];
  const int b = blockIdx.x / M2n, ky = blockIdx.x % M2n;
  const int tid = threadIdx.x, wid = tid >> 5, lane = tid & 31;

  for (int idx = tid; idx < 4096; idx += 128) {
    int c = idx >> 6, k = idx & 63;
    int kxm = k & 31, half = k >> 5;
    float val = 0.f;
    if (kxm < M1n)
      val = (half ? m2_im : m2_re)[(((size_t)b * M1n + kxm) * M2n + ky) * Cn + c];
    ldsBT[c][k] = (_Float16)val;
  }
  for (int idx = tid; idx < 16384; idx += 128) {
    int hrow = idx >> 6, k = idx & 63;
    int kxm = k & 31, half = k >> 5;
    float vre = 0.f, vim = 0.f;
    if (kxm < M1n) {
      int f = kxm - 10;
      int r = (f * hrow) & 255;
      float a = (float)r * (1.0f / 128.0f);
      float c0 = cospif(a), s0 = sinpif(a);
      vre = half ? -s0 : c0;   // g_re = cos*m2_re - sin*m2_im
      vim = half ? c0 : s0;    // g_im = sin*m2_re + cos*m2_im
    }
    ldsAre[hrow][k] = (_Float16)vre;
    ldsAim[hrow][k] = (_Float16)vim;
  }
  __syncthreads();
  v16h b0 = frag_b(&ldsBT[wid * 16][0], 64, lane);
  v16h b1 = frag_b(&ldsBT[wid * 16][32], 64, lane);
  const int lg = lane >> 4, nl = lane & 15;
  const int c = wid * 16 + nl;
  for (int mt = 0; mt < 16; ++mt) {
    v8f accRe = {}, accIm = {};
    accRe = wmma16(frag_a(&ldsAre[mt * 16][0], 64, lane), b0, accRe);
    accIm = wmma16(frag_a(&ldsAim[mt * 16][0], 64, lane), b0, accIm);
    accRe = wmma16(frag_a(&ldsAre[mt * 16][32], 64, lane), b1, accRe);
    accIm = wmma16(frag_a(&ldsAim[mt * 16][32], 64, lane), b1, accIm);
    for (int e = 0; e < 8; ++e) {
      int h = mt * 16 + e + 8 * lg;
      size_t o = ((size_t)(b * Hn + h) * M2n + ky) * Cn + c;
      g_re[o] = accRe[e] * (1.0f / 256.0f);   // 1/H
      g_im[o] = accIm[e] * (1.0f / 256.0f);
    }
  }
}

// ---------------------------------------------------------------------------
// Stage 5: fused irfft along W (incl. 1/W, Hermitian factor 2, Im(X0) dropped)
//          + points @ w_lin^T + residual.   One workgroup per (b,h).
// GEMM accumulate: O(256x64) = T(256x64)@G(64x64) + P(256x64)@Wl^T(64x64), +P.
// ---------------------------------------------------------------------------
__global__ __launch_bounds__(128) void k5_out(const float* __restrict__ g_re,
                                              const float* __restrict__ g_im,
                                              const float* __restrict__ points,
                                              const float* __restrict__ w_lin,
                                              float* __restrict__ out) {
  __shared__ __align__(64) _Float16 ldsT[256][64];   // irfft coeff matrix
  __shared__ __align__(64) _Float16 ldsP[256][64];   // points tile (f16 A)
  __shared__ __align__(64) _Float16 ldsG[64][64];    // [g_re|g_im]^T  [c][k]
  __shared__ __align__(64) _Float16 ldsWl[64][64];   // w_lin [c][j]
  const int bh = blockIdx.x;
  const int tid = threadIdx.x, wid = tid >> 5, lane = tid & 31;
  const size_t pbase = (size_t)bh * Wn * Cn;

  for (int idx = tid; idx < 16384; idx += 128) {
    int w = idx >> 6, k = idx & 63;
    int kym = k & 31, half = k >> 5;
    float val = 0.f;
    if (kym < M2n) {
      if (kym == 0) {
        val = half ? 0.f : (1.0f / 256.0f);   // DC: Re only, no factor 2
      } else {
        int r = (kym * w) & 255;
        float a = (float)r * (1.0f / 128.0f);
        val = half ? (-2.0f / 256.0f) * sinpif(a)
                   : (2.0f / 256.0f) * cospif(a);
      }
    }
    ldsT[w][k] = (_Float16)val;
  }
  for (int idx = tid; idx < 16384; idx += 128) {
    int w = idx >> 6, j = idx & 63;
    ldsP[w][j] = (_Float16)points[pbase + (size_t)w * Cn + j];
  }
  for (int idx = tid; idx < 4096; idx += 128) {
    int c = idx >> 6, k = idx & 63;
    int kym = k & 31, half = k >> 5;
    float val = 0.f;
    if (kym < M2n)
      val = (half ? g_im : g_re)[((size_t)bh * M2n + kym) * Cn + c];
    ldsG[c][k] = (_Float16)val;
  }
  for (int idx = tid; idx < 4096; idx += 128) {
    int c = idx >> 6, j = idx & 63;
    ldsWl[c][j] = (_Float16)w_lin[c * Cn + j];   // (w_lin^T)[j][c] transposed
  }
  __syncthreads();
  v16h bg0 = frag_b(&ldsG[wid * 16][0], 64, lane);
  v16h bg1 = frag_b(&ldsG[wid * 16][32], 64, lane);
  v16h bw0 = frag_b(&ldsWl[wid * 16][0], 64, lane);
  v16h bw1 = frag_b(&ldsWl[wid * 16][32], 64, lane);
  const int lg = lane >> 4, nl = lane & 15;
  const int c = wid * 16 + nl;
  for (int mt = 0; mt < 16; ++mt) {
    v8f acc = {};
    acc = wmma16(frag_a(&ldsT[mt * 16][0], 64, lane), bg0, acc);
    acc = wmma16(frag_a(&ldsT[mt * 16][32], 64, lane), bg1, acc);
    acc = wmma16(frag_a(&ldsP[mt * 16][0], 64, lane), bw0, acc);
    acc = wmma16(frag_a(&ldsP[mt * 16][32], 64, lane), bw1, acc);
    for (int e = 0; e < 8; ++e) {
      int w = mt * 16 + e + 8 * lg;
      size_t o = pbase + (size_t)w * Cn + c;
      out[o] = acc[e] + points[o];   // residual in f32
    }
  }
}

// ---------------------------------------------------------------------------
extern "C" void kernel_launch(void* const* d_in, const int* in_sizes, int n_in,
                              void* d_out, int out_size, void* d_ws,
                              size_t ws_size, hipStream_t stream) {
  (void)in_sizes; (void)n_in; (void)out_size; (void)ws_size;
  const float* points = (const float*)d_in[0];
  const float* ker_re = (const float*)d_in[1];
  const float* ker_im = (const float*)d_in[2];
  const float* w_lin  = (const float*)d_in[3];
  float* out = (float*)d_out;
  float* ws  = (float*)d_ws;

  constexpr size_t TSZ = (size_t)Bn * Hn * M2n * Cn;   // 5,242,880 floats
  constexpr size_t MSZ = (size_t)Bn * M1n * M2n * Cn;  //   409,600 floats
  // Total workspace: 2*TSZ + 4*MSZ floats = ~48.5 MB (t/g buffers aliased).
  float* t_re  = ws;
  float* t_im  = ws + TSZ;
  float* m_re  = ws + 2 * TSZ;
  float* m_im  = ws + 2 * TSZ + MSZ;
  float* m2_re = ws + 2 * TSZ + 2 * MSZ;
  float* m2_im = ws + 2 * TSZ + 3 * MSZ;
  float* g_re  = t_re;   // t dead after stage 2; reuse for g
  float* g_im  = t_im;

  k1_dft_w<<<Bn * Hn, 128, 0, stream>>>(points, t_re, t_im);
  k2_dft_h<<<Bn * M2n, 128, 0, stream>>>(t_re, t_im, m_re, m_im);
  k3_modes<<<M1n * M2n, 128, 0, stream>>>(m_re, m_im, ker_re, ker_im, m2_re, m2_im);
  k4_idft_h<<<Bn * M2n, 128, 0, stream>>>(m2_re, m2_im, g_re, g_im);
  k5_out<<<Bn * Hn, 128, 0, stream>>>(g_re, g_im, points, w_lin, out);
}
